// Linear_Attention_78245714198568
// MI455X (gfx1250) — matrix operand order repaired and checked by reading, not compiled, not run on hardware
//
#include <hip/hip_runtime.h>

// ---------------------------------------------------------------------------
// Linear attention for MI455X (gfx1250), wave32, WMMA bf16 pipeline.
//   B=4, L=S=4096, CH=512, H=8, DH=64
// Pipeline (all GEMMs via v_wmma_f32_16x16x32_bf16, fp32 accumulate):
//   1) qb = sigmoid(queries @ Wq^T + bq)  -> bf16   [16384,512]
//      kb = sigmoid(keys    @ Wk^T + bk)  -> bf16
//      vb =         values  @ Wv^T + bv   -> bf16
//   2) per (b,h): KV[64,64] = k^T @ v (K=4096), ksum[64] = sum_s k
//   3) per (b,h,ltile): x = (q @ KV) * (1/(q . ksum + 1e-6))  -> bf16
//   4) out = (x @ Wo^T + bo) * mask  -> fp32
// bf16-only LDS staging uses global_load_async_to_lds_b128 (ASYNCcnt path).
// ---------------------------------------------------------------------------

typedef unsigned short u16;
typedef __bf16 v16bf __attribute__((ext_vector_type(16)));
typedef float  v8f   __attribute__((ext_vector_type(8)));

__device__ __forceinline__ u16 f2bf(float f) {
  unsigned int u = __float_as_uint(f);
  u += 0x7FFFu + ((u >> 16) & 1u);   // round-to-nearest-even
  return (u16)(u >> 16);
}
__device__ __forceinline__ float bf2f(u16 x) {
  return __uint_as_float(((unsigned int)x) << 16);
}

__device__ __forceinline__ v8f wmma_bf16(v16bf a, v16bf b, v8f c) {
  return __builtin_amdgcn_wmma_f32_16x16x32_bf16(
      /*neg_a=*/false, a, /*neg_b=*/false, b,
      /*c_mod=*/(short)0, c, /*reuse_a=*/false, /*reuse_b=*/false);
}

// Async copy of 16 bytes global -> LDS (per active lane). LDS destination is
// the low 32 bits of the generic pointer (ISA 10.2: LDS addr = addr[31:0]).
__device__ __forceinline__ void async_copy_b128(void* lds_dst, const void* gsrc) {
  const unsigned lds = (unsigned)(unsigned long long)lds_dst;
  const unsigned long long ga = (unsigned long long)gsrc;
  asm volatile("global_load_async_to_lds_b128 %0, %1, off"
               :: "v"(lds), "v"(ga) : "memory");
}
__device__ __forceinline__ void async_wait0() {
  asm volatile("s_wait_asynccnt 0" ::: "memory");
}

// A fragment: 16x32 (MxK) bf16. Per ISA: lane = g*16+r holds row M=r;
// VGPR j holds K pair at (j<4?0:16) + g*8 + (j&3)*2. sA is [M][ldr] bf16.
__device__ __forceinline__ v16bf fragA(const u16* sA, int ldr, int mBase,
                                       int k0, int lane) {
  const int r = lane & 15, g = lane >> 4;
  const u16* p = sA + (size_t)(mBase + r) * ldr + k0;
  union { v16bf v; unsigned int u[8]; } f;
#pragma unroll
  for (int j = 0; j < 8; ++j) {
    const int kb = ((j < 4) ? 0 : 16) + g * 8 + (j & 3) * 2;
    f.u[j] = *(const unsigned int*)(p + kb);
  }
  return f.v;
}

// B fragment: 32x16 (KxN) bf16. lane = g*16+c holds column N=c;
// VGPR j holds K pair at (j<4?0:16) + g*8 + (j&3)*2. sB is stored N-major: [N][ldr] bf16.
__device__ __forceinline__ v16bf fragB(const u16* sB, int ldr, int nBase,
                                       int k0, int lane) {
  const int c = lane & 15, g = lane >> 4;
  const u16* p = sB + (size_t)(nBase + c) * ldr + k0;
  union { v16bf v; unsigned int u[8]; } f;
#pragma unroll
  for (int j = 0; j < 8; ++j) f.u[j] = *(const unsigned int*)(p + ((j < 4) ? 0 : 16) + g * 8 + (j & 3) * 2);
  return f.v;
}

// ---------------------------------------------------------------------------
// Kernel 1/4: C[M,512] = act(A[M,512] @ W[512,512]^T + bias)  (W row-major
// [n][k] == B fragment layout directly). BM=128, BN=128, BK=32 (one WMMA K).
// 8 waves as 4x2 grid of 32x64 sub-tiles -> 8 WMMA per wave per K-step.
// ABF16: A is bf16 (async-to-LDS staging). SIG: sigmoid. MASKOUT: fp32*mask.
// ---------------------------------------------------------------------------
template <bool ABF16, bool SIG, bool MASKOUT>
__global__ __launch_bounds__(256) void gemm512_kernel(
    const void* __restrict__ Ain, const float* __restrict__ W,
    const float* __restrict__ bias, u16* __restrict__ Cb,
    float* __restrict__ Cf, const float* __restrict__ mask) {
  constexpr int BM = 128, BN = 128, BK = 32, LDR = BK + 8;
  __shared__ u16 sA[BM * LDR];
  __shared__ u16 sB[BN * LDR];
  const int tid = threadIdx.x;
  const int lane = tid & 31;
  const int wave = tid >> 5;                 // 8 waves
  const int m0 = blockIdx.x * BM;
  const int n0 = blockIdx.y * BN;
  const int wm = (wave & 3) * 32;            // 4x2 wave grid, 32x64 per wave
  const int wn = (wave >> 2) * 64;

  v8f acc[2][4] = {};

  for (int k0 = 0; k0 < 512; k0 += BK) {
    __syncthreads();
    if (ABF16) {
      const u16* A = (const u16*)Ain;
      for (int it = tid; it < BM * BK / 8; it += 256) {
        const int row = it >> 2, c8 = (it & 3) * 8;
        async_copy_b128(&sA[row * LDR + c8],
                        &A[(size_t)(m0 + row) * 512 + k0 + c8]);
      }
      if (k0 + BK < 512)  // gfx1250 global_prefetch of next K slab
        __builtin_prefetch(&A[(size_t)(m0 + (tid & 127)) * 512 + k0 + BK], 0, 1);
    } else {
      const float* A = (const float*)Ain;
      for (int it = tid; it < BM * BK / 4; it += 256) {
        const int row = it >> 3, c4 = (it & 7) * 4;
        const float4 f = *(const float4*)&A[(size_t)(m0 + row) * 512 + k0 + c4];
        const unsigned int p0 = f2bf(f.x) | ((unsigned int)f2bf(f.y) << 16);
        const unsigned int p1 = f2bf(f.z) | ((unsigned int)f2bf(f.w) << 16);
        *(unsigned int*)&sA[row * LDR + c4] = p0;
        *(unsigned int*)&sA[row * LDR + c4 + 2] = p1;
      }
      if (k0 + BK < 512)
        __builtin_prefetch(&A[(size_t)(m0 + (tid & 127)) * 512 + k0 + BK], 0, 1);
    }
    for (int it = tid; it < BN * BK / 4; it += 256) {
      const int row = it >> 3, c4 = (it & 7) * 4;
      const float4 f = *(const float4*)&W[(size_t)(n0 + row) * 512 + k0 + c4];
      const unsigned int p0 = f2bf(f.x) | ((unsigned int)f2bf(f.y) << 16);
      const unsigned int p1 = f2bf(f.z) | ((unsigned int)f2bf(f.w) << 16);
      *(unsigned int*)&sB[row * LDR + c4] = p0;
      *(unsigned int*)&sB[row * LDR + c4 + 2] = p1;
    }
    if (ABF16) async_wait0();
    __syncthreads();

#pragma unroll
    for (int mt = 0; mt < 2; ++mt) {
      const v16bf a = fragA(sA, LDR, wm + mt * 16, 0, lane);
#pragma unroll
      for (int nt = 0; nt < 4; ++nt) {
        const v16bf b = fragB(sB, LDR, wn + nt * 16, 0, lane);
        acc[mt][nt] = wmma_bf16(a, b, acc[mt][nt]);
      }
    }
  }

  // Epilogue. C layout: VGPR rv, lanes0-15 -> M=rv, lanes16-31 -> M=8+rv; N=lane&15.
  const int r = lane & 15, g = lane >> 4;
#pragma unroll
  for (int mt = 0; mt < 2; ++mt) {
#pragma unroll
    for (int nt = 0; nt < 4; ++nt) {
      const int n = n0 + wn + nt * 16 + r;
      const float bn = bias[n];
#pragma unroll
      for (int rv = 0; rv < 8; ++rv) {
        const int m = m0 + wm + mt * 16 + g * 8 + rv;
        float v = acc[mt][nt][rv] + bn;
        if (SIG) v = 1.0f / (1.0f + __expf(-v));
        if (MASKOUT) {
          Cf[(size_t)m * 512 + n] = v * mask[m];   // mask flat [B*L]
        } else {
          Cb[(size_t)m * 512 + n] = f2bf(v);
        }
      }
    }
  }
}

// ---------------------------------------------------------------------------
// Kernel 2: per (b,h): KV[64,64] = k^T @ v over s=4096; ksum[64] = sum_s k.
// k,v staged transposed into LDS ([d][s] / [e][s]) so fragments read
// contiguous K pairs. 32 blocks, 8 waves, 2 output tiles per wave.
// (Transpose staging cannot use async copies: 2-byte scatter.)
// ---------------------------------------------------------------------------
__global__ __launch_bounds__(256) void kv_ksum_kernel(
    const u16* __restrict__ kq, const u16* __restrict__ vq,
    float* __restrict__ KV, float* __restrict__ ksum) {
  constexpr int SC = 64, LDR = SC + 8;
  __shared__ u16 sK[64 * LDR];
  __shared__ u16 sV[64 * LDR];
  const int tid = threadIdx.x, lane = tid & 31, wave = tid >> 5;
  const int bh = blockIdx.x, b = bh >> 3, h = bh & 7;
  const u16* kbase = kq + (size_t)b * 4096 * 512 + h * 64;
  const u16* vbase = vq + (size_t)b * 4096 * 512 + h * 64;

  v8f acc[2] = {};
  const int t0 = wave * 2, t1 = wave * 2 + 1;  // 16 tiles = 4x4 of 16x16
  float ks = 0.0f;

  for (int s0 = 0; s0 < 4096; s0 += SC) {
    __syncthreads();
    for (int it = tid; it < 512; it += 256) {   // 64 s x 8 d-octets
      const int sl = it >> 3, d0 = (it & 7) * 8;
      const uint4 kd = *(const uint4*)&kbase[(size_t)(s0 + sl) * 512 + d0];
      const uint4 vd = *(const uint4*)&vbase[(size_t)(s0 + sl) * 512 + d0];
      const u16* kp = (const u16*)&kd;
      const u16* vp = (const u16*)&vd;
#pragma unroll
      for (int i = 0; i < 8; ++i) {             // transpose into LDS
        sK[(d0 + i) * LDR + sl] = kp[i];
        sV[(d0 + i) * LDR + sl] = vp[i];
      }
    }
    if (s0 + SC < 4096)
      __builtin_prefetch(&kbase[(size_t)(s0 + SC + (tid >> 2)) * 512], 0, 1);
    __syncthreads();

#pragma unroll
    for (int ks0 = 0; ks0 < SC; ks0 += 32) {
      {
        const int mt = t0 >> 2, nt = t0 & 3;
        const v16bf a = fragA(sK, LDR, mt * 16, ks0, lane);
        const v16bf bb = fragB(sV, LDR, nt * 16, ks0, lane);
        acc[0] = wmma_bf16(a, bb, acc[0]);
      }
      {
        const int mt = t1 >> 2, nt = t1 & 3;
        const v16bf a = fragA(sK, LDR, mt * 16, ks0, lane);
        const v16bf bb = fragB(sV, LDR, nt * 16, ks0, lane);
        acc[1] = wmma_bf16(a, bb, acc[1]);
      }
    }
    if (tid < 64) {                             // fused k row-sum
      float s = 0.0f;
#pragma unroll 8
      for (int i = 0; i < SC; ++i) s += bf2f(sK[tid * LDR + i]);
      ks += s;
    }
  }

  float* kvp = KV + (size_t)bh * 4096;
  const int r = lane & 15, g = lane >> 4;
#pragma unroll
  for (int i = 0; i < 2; ++i) {
    const int t = (i == 0) ? t0 : t1;
    const int mt = t >> 2, nt = t & 3;
    const int e = nt * 16 + r;
#pragma unroll
    for (int rv = 0; rv < 8; ++rv) {
      const int d = mt * 16 + g * 8 + rv;
      kvp[d * 64 + e] = acc[i][rv];
    }
  }
  if (tid < 64) ksum[bh * 64 + tid] = ks;
}

// ---------------------------------------------------------------------------
// Kernel 3: per (b,h,128-row tile): x = (q @ KV) * Z, Z = 1/(q.ksum + 1e-6).
// q tile staged via async-to-LDS; KV transposed+converted on the VALU path.
// ---------------------------------------------------------------------------
__global__ __launch_bounds__(256) void attn_out_kernel(
    const u16* __restrict__ qq, const float* __restrict__ KV,
    const float* __restrict__ ksum, u16* __restrict__ xb) {
  constexpr int LDR = 72;
  __shared__ u16 sQ[128 * LDR];
  __shared__ u16 sKV[64 * LDR];   // transposed: [e][d]
  __shared__ float sks[64];
  __shared__ float sz[128];
  const int tid = threadIdx.x, lane = tid & 31, wave = tid >> 5;
  const int l0 = blockIdx.x * 128;
  const int h = blockIdx.y, b = blockIdx.z;
  const int bh = b * 8 + h;
  const u16* qbase = qq + (size_t)b * 4096 * 512 + h * 64;
  const float* kvp = KV + (size_t)bh * 4096;

  for (int it = tid; it < 1024; it += 256) {    // 128 rows x 8 octets of q
    const int row = it >> 3, c8 = (it & 7) * 8;
    async_copy_b128(&sQ[row * LDR + c8],
                    &qbase[(size_t)(l0 + row) * 512 + c8]);
  }
  for (int it = tid; it < 4096; it += 256) {    // KV fp32 [d][e] -> bf16 [e][d]
    const int d = it >> 6, e = it & 63;
    sKV[e * LDR + d] = f2bf(kvp[d * 64 + e]);
  }
  if (tid < 64) sks[tid] = ksum[bh * 64 + tid];
  async_wait0();
  __syncthreads();
  if (tid < 128) {                              // normalizer per row
    float s = 0.0f;
#pragma unroll 8
    for (int d = 0; d < 64; ++d) s += bf2f(sQ[tid * LDR + d]) * sks[d];
    sz[tid] = 1.0f / (s + 1e-6f);
  }
  __syncthreads();

  v8f acc[4] = {};
#pragma unroll
  for (int i = 0; i < 4; ++i) {                 // 32 tiles (8x4) over 8 waves
    const int t = wave * 4 + i, mt = t >> 2, nt = t & 3;
#pragma unroll
    for (int k0 = 0; k0 < 64; k0 += 32) {
      const v16bf a = fragA(sQ, LDR, mt * 16, k0, lane);
      const v16bf bb = fragB(sKV, LDR, nt * 16, k0, lane);
      acc[i] = wmma_bf16(a, bb, acc[i]);
    }
  }

  const int r = lane & 15, g = lane >> 4;
#pragma unroll
  for (int i = 0; i < 4; ++i) {
    const int t = wave * 4 + i, mt = t >> 2, nt = t & 3;
#pragma unroll
    for (int rv = 0; rv < 8; ++rv) {
      const int ml = mt * 16 + g * 8 + rv;
      const float v = acc[i][rv] * sz[ml];
      xb[(size_t)(b * 4096 + l0 + ml) * 512 + h * 64 + nt * 16 + r] = f2bf(v);
    }
  }
}

// ---------------------------------------------------------------------------
extern "C" void kernel_launch(void* const* d_in, const int* in_sizes, int n_in,
                              void* d_out, int out_size, void* d_ws,
                              size_t ws_size, hipStream_t stream) {
  const float* queries = (const float*)d_in[0];
  const float* keys    = (const float*)d_in[1];
  const float* values  = (const float*)d_in[2];
  const float* mask    = (const float*)d_in[3];
  const float* Wq = (const float*)d_in[4];
  const float* bq = (const float*)d_in[5];
  const float* Wk = (const float*)d_in[6];
  const float* bk = (const float*)d_in[7];
  const float* Wv = (const float*)d_in[8];
  const float* bv = (const float*)d_in[9];
  const float* Wo = (const float*)d_in[10];
  const float* bo = (const float*)d_in[11];

  const size_t elems = (size_t)4 * 4096 * 512;  // B*L*CH = 8,388,608
  char* w = (char*)d_ws;
  u16* qb = (u16*)w;  w += elems * 2;
  u16* kb = (u16*)w;  w += elems * 2;
  u16* vb = (u16*)w;  w += elems * 2;
  u16* xb = (u16*)w;  w += elems * 2;
  float* KV = (float*)w;  w += (size_t)32 * 64 * 64 * 4;
  float* ks = (float*)w;  w += (size_t)32 * 64 * 4;

  const dim3 gp(16384 / 128, 512 / 128);  // (128, 4)

  // Projections (sigmoid on q,k)
  gemm512_kernel<false, true, false><<<gp, 256, 0, stream>>>(
      queries, Wq, bq, qb, nullptr, nullptr);
  gemm512_kernel<false, true, false><<<gp, 256, 0, stream>>>(
      keys, Wk, bk, kb, nullptr, nullptr);
  gemm512_kernel<false, false, false><<<gp, 256, 0, stream>>>(
      values, Wv, bv, vb, nullptr, nullptr);

  // KV accumulation + k sums
  kv_ksum_kernel<<<32, 256, 0, stream>>>(kb, vb, KV, ks);

  // q @ KV with fused normalizer
  attn_out_kernel<<<dim3(32, 8, 4), 256, 0, stream>>>(qb, KV, ks, xb);

  // Output projection + mask (fp32 out)
  gemm512_kernel<true, false, true><<<gp, 256, 0, stream>>>(
      xb, Wo, bo, nullptr, (float*)d_out, mask);
}
